// SimpleRNN_51453708206132
// MI455X (gfx1250) — matrix operand order repaired and checked by reading, not compiled, not run on hardware
//
#include <hip/hip_runtime.h>

// ---------------------------------------------------------------------------
// Elman RNN for MI455X (gfx1250, wave32, WMMA, async global->LDS).
//   x_proj = x @ W_ih^T + b_ih + b_hh   (bf16 WMMA GEMM, async-LDS double buffer)
//   h_t    = tanh(x_proj_t + h_{t-1} @ W_hh^T)  (persistent kernel, grid barrier)
//   out    = h_S @ W_ho^T + b_ho
// ---------------------------------------------------------------------------

constexpr int NBATCH  = 64;
constexpr int NSEQ    = 512;
constexpr int DIM_IN  = 512;
constexpr int DIM_HID = 1024;
constexpr int DIM_OUT = 512;

typedef __attribute__((ext_vector_type(16))) __bf16 bf16x16;
typedef __attribute__((ext_vector_type(8)))  __bf16 bf16x8;
typedef __attribute__((ext_vector_type(8)))  float  f32x8;
typedef int v4i __attribute__((vector_size(16)));

// CDNA5 async global->LDS copy path (ASYNCcnt-tracked), guarded so the file
// compiles on toolchains lacking the builtins.
#if defined(__AMDGCN__) &&                                                   \
    __has_builtin(__builtin_amdgcn_global_load_async_to_lds_b128) &&         \
    __has_builtin(__builtin_amdgcn_s_wait_asynccnt)
#define HAVE_ASYNC_LDS 1
#else
#define HAVE_ASYNC_LDS 0
#endif

// ---- WMMA: D = A(16x32 bf16) * B(32x16 bf16) + C(16x16 f32) ---------------
__device__ __forceinline__ f32x8 wmma_bf16(bf16x16 a, bf16x16 b, f32x8 c) {
  return __builtin_amdgcn_wmma_f32_16x16x32_bf16(false, a, false, b,
                                                 (short)0, c, false, false);
}

__device__ __forceinline__ bf16x16 cat8(bf16x8 lo, bf16x8 hi) {
  return __builtin_shufflevector(lo, hi, 0, 1, 2, 3, 4, 5, 6, 7,
                                         8, 9, 10, 11, 12, 13, 14, 15);
}

// A fragment (16x32 bf16), row-major source with leading dim ld.
// lane<16: row=lane, k={0..7,16..23}; lane>=16: row=lane-16, k={8..15,24..31}.
__device__ __forceinline__ bf16x16 load_a_frag(const __bf16* base, int ld, int lane) {
  const int half = lane >> 4, m = lane & 15;
  const __bf16* p = base + (size_t)m * ld + half * 8;
  bf16x8 lo = *(const bf16x8*)(p);
  bf16x8 hi = *(const bf16x8*)(p + 16);
  return cat8(lo, hi);
}

// B fragment (32x16 bf16) from row-major W[N, K].
// lane<16: n=lane, k={0..7,16..23}; lane>=16: n=lane-16, k={8..15,24..31}.
__device__ __forceinline__ bf16x16 load_b_frag(const __bf16* base, int ld, int lane) {
  const int half = lane >> 4, n = lane & 15;
  const __bf16* p = base + (size_t)n * ld + half * 8;
  bf16x8 lo = *(const bf16x8*)(p);
  bf16x8 hi = *(const bf16x8*)(p + 16);
  return cat8(lo, hi);
}

// 16-byte async (or fallback sync) global -> LDS copy.
__device__ __forceinline__ void copy16_to_lds(const __bf16* g, __bf16* l) {
#if HAVE_ASYNC_LDS
  __builtin_amdgcn_global_load_async_to_lds_b128(
      (__attribute__((address_space(1))) v4i*)(__bf16*)g,
      (__attribute__((address_space(3))) v4i*)l, 0, 0);
#else
  *(bf16x8*)l = *(const bf16x8*)g;
#endif
}

__device__ __forceinline__ void wait_async_done() {
#if HAVE_ASYNC_LDS
  __builtin_amdgcn_s_wait_asynccnt(0);
#endif
}

// ---------------------------------------------------------------------------
// Vectorized fp32 -> bf16 convert: 8 elements/thread (2x b128 load, 1x b128 store)
// ---------------------------------------------------------------------------
__global__ void k_f32_to_bf16v(const float* __restrict__ src,
                               __bf16* __restrict__ dst, int n8) {
  int i = blockIdx.x * blockDim.x + threadIdx.x;
  if (i < n8) {
    float4 a = ((const float4*)src)[2 * i];
    float4 b = ((const float4*)src)[2 * i + 1];
    bf16x8 v;
    v[0] = (__bf16)a.x; v[1] = (__bf16)a.y; v[2] = (__bf16)a.z; v[3] = (__bf16)a.w;
    v[4] = (__bf16)b.x; v[5] = (__bf16)b.y; v[6] = (__bf16)b.z; v[7] = (__bf16)b.w;
    ((bf16x8*)dst)[i] = v;
  }
}

// ---------------------------------------------------------------------------
// GEMM1: x_proj[s, b, h] = sum_k x[b,s,k] * W_ih[h,k] + b_ih[h] + b_hh[h]
// All-bf16 inputs. Block 256 threads (8 waves = 4m x 2n, 32x32 C per wave).
// Block tile 128(M) x 64(N); K in 32-chunks, double-buffered in LDS via
// async global->LDS copies overlapping the WMMAs of the previous chunk.
// Output stored TRANSPOSED as [SEQ, BATCH, HID] so the scan reads a
// contiguous 256 KB slice per timestep.
// ---------------------------------------------------------------------------
__global__ void k_xproj_gemm(const __bf16* __restrict__ xbf,
                             const __bf16* __restrict__ Wihbf,
                             const float* __restrict__ b_ih,
                             const float* __restrict__ b_hh,
                             float* __restrict__ xproj) {
  constexpr int MT = 128, NT = 64, KT = 32, LD = KT + 8;  // LDS pad vs banks
  constexpr int NK = DIM_IN / KT;                         // 16
  __shared__ __bf16 As[2][MT * LD];
  __shared__ __bf16 Bs[2][NT * LD];

  const int tid  = threadIdx.x;
  const int lane = tid & 31;
  const int wv   = tid >> 5;
  const int wm   = wv & 3;    // 0..3 -> M
  const int wn   = wv >> 2;   // 0..1 -> N
  const int m_base = blockIdx.x * MT;
  const int n_base = blockIdx.y * NT;

  const __bf16* gA = xbf   + (size_t)m_base * DIM_IN;
  const __bf16* gB = Wihbf + (size_t)n_base * DIM_IN;

  // Per-chunk staging: A = 512 16B-chunks (2/thread), B = 256 (1/thread).
  const int qa_row0 = tid >> 2, qa_col = (tid & 3) * 8;       // A chunk 0
  const int qa_row1 = (tid + 256) >> 2;                       // A chunk 1
  const int qb_row  = tid >> 2, qb_col = (tid & 3) * 8;       // B chunk

  auto stage = [&](int k0, int buf) {
    copy16_to_lds(gA + (size_t)qa_row0 * DIM_IN + k0 + qa_col,
                  &As[buf][qa_row0 * LD + qa_col]);
    copy16_to_lds(gA + (size_t)qa_row1 * DIM_IN + k0 + qa_col,
                  &As[buf][qa_row1 * LD + qa_col]);
    copy16_to_lds(gB + (size_t)qb_row * DIM_IN + k0 + qb_col,
                  &Bs[buf][qb_row * LD + qb_col]);
  };

  f32x8 c[2][2] = {};

  stage(0, 0);
  wait_async_done();
  __syncthreads();

  for (int kc = 0; kc < NK; ++kc) {
    const int cur = kc & 1;
    if (kc + 1 < NK) stage((kc + 1) * KT, cur ^ 1);  // overlap with WMMAs

    bf16x16 a0 = load_a_frag(&As[cur][(wm * 32 + 0)  * LD], LD, lane);
    bf16x16 a1 = load_a_frag(&As[cur][(wm * 32 + 16) * LD], LD, lane);
    bf16x16 b0 = load_b_frag(&Bs[cur][(wn * 32 + 0)  * LD], LD, lane);
    bf16x16 b1 = load_b_frag(&Bs[cur][(wn * 32 + 16) * LD], LD, lane);
    c[0][0] = wmma_bf16(a0, b0, c[0][0]);
    c[0][1] = wmma_bf16(a0, b1, c[0][1]);
    c[1][0] = wmma_bf16(a1, b0, c[1][0]);
    c[1][1] = wmma_bf16(a1, b1, c[1][1]);

    if (kc + 1 < NK) wait_async_done();  // next chunk's copies landed
    __syncthreads();                     // all waves done reading `cur`
  }

  // Epilogue: +bias, store transposed [S, B, H]. Row m = b*NSEQ + s.
  const int half = lane >> 4, nl = lane & 15;
#pragma unroll
  for (int i = 0; i < 2; ++i) {
#pragma unroll
    for (int j = 0; j < 2; ++j) {
      const int col  = n_base + wn * 32 + j * 16 + nl;
      const float bias = b_ih[col] + b_hh[col];
      const int rbase = m_base + wm * 32 + i * 16 + half * 8;
#pragma unroll
      for (int r = 0; r < 8; ++r) {
        const int row = rbase + r;                 // b*NSEQ + s
        const int b = row >> 9, s = row & (NSEQ - 1);
        xproj[((size_t)s * NBATCH + b) * DIM_HID + col] = c[i][j][r] + bias;
      }
    }
  }
}

// ---------------------------------------------------------------------------
// Persistent recurrent scan: 64 blocks x 128 threads = 256 waves; wave gw owns
// the 16x16 tile (m0 = (gw&3)*16 of batch, n0 = (gw>>2)*16 of hidden).
// Device-wide barrier between steps via monotone atomic counter.
// x_proj layout is [S, B, H]: each step reads a contiguous slice.
// ---------------------------------------------------------------------------
__global__ void k_rnn_scan(const float* __restrict__ xproj,
                           const __bf16* __restrict__ Whh,
                           __bf16* __restrict__ h0,
                           __bf16* __restrict__ h1,
                           unsigned* __restrict__ bar) {
  const int tid  = threadIdx.x;
  const int lane = tid & 31;
  const int gw   = blockIdx.x * 4 + (tid >> 5);
  const int m0   = (gw & 3) * 16;
  const int n0   = (gw >> 2) * 16;
  const int half = lane >> 4, nl = lane & 15;
  const int col  = n0 + nl;

  __bf16* hc = h0;  // zero-initialized by host-side memsetAsync
  __bf16* hn = h1;
  unsigned gen = 0;

  for (int t = 0; t < NSEQ; ++t) {
    f32x8 c = {};
#pragma unroll 4
    for (int k0 = 0; k0 < DIM_HID; k0 += 32) {
      bf16x16 a = load_a_frag(hc + (size_t)m0 * DIM_HID + k0, DIM_HID, lane);
      bf16x16 b = load_b_frag(Whh + (size_t)n0 * DIM_HID + k0, DIM_HID, lane);
      c = wmma_bf16(a, b, c);
    }
    const float* xp = xproj + (size_t)t * NBATCH * DIM_HID;
#pragma unroll
    for (int r = 0; r < 8; ++r) {
      const int brow = m0 + half * 8 + r;
      const float v = c[r] + xp[(size_t)brow * DIM_HID + col];
      hn[(size_t)brow * DIM_HID + col] = (__bf16)tanhf(v);
    }

    // ---- device-wide barrier (release h writes, acquire for next step) ----
    __threadfence();
    __syncthreads();
    ++gen;
    if (tid == 0) {
      atomicAdd(bar, 1u);
      while (__hip_atomic_load(bar, __ATOMIC_ACQUIRE, __HIP_MEMORY_SCOPE_AGENT)
             < gen * gridDim.x) { /* spin */ }
    }
    __syncthreads();
    __threadfence();

    __bf16* tmp = hc; hc = hn; hn = tmp;
  }
  // 512 steps: t=511 (odd) wrote h0 -> final hidden state lives in h0.
}

// ---------------------------------------------------------------------------
// Output projection: out[b, o] = h_final[b, :] . W_ho[o, :] + b_ho[o]
// 32 blocks x 128 threads = 128 waves = 4(M) x 32(N) 16x16 tiles.
// ---------------------------------------------------------------------------
__global__ void k_out_proj(const __bf16* __restrict__ hfin,
                           const __bf16* __restrict__ Who,
                           const float* __restrict__ b_ho,
                           float* __restrict__ out) {
  const int tid  = threadIdx.x;
  const int lane = tid & 31;
  const int gw   = blockIdx.x * 4 + (tid >> 5);
  const int m0   = (gw & 3) * 16;
  const int n0   = (gw >> 2) * 16;
  const int half = lane >> 4, nl = lane & 15;

  f32x8 c = {};
#pragma unroll 4
  for (int k0 = 0; k0 < DIM_HID; k0 += 32) {
    bf16x16 a = load_a_frag(hfin + (size_t)m0 * DIM_HID + k0, DIM_HID, lane);
    bf16x16 b = load_b_frag(Who + (size_t)n0 * DIM_HID + k0, DIM_HID, lane);
    c = wmma_bf16(a, b, c);
  }
  const int col = n0 + nl;
  const float bias = b_ho[col];
#pragma unroll
  for (int r = 0; r < 8; ++r) {
    const int brow = m0 + half * 8 + r;
    out[(size_t)brow * DIM_OUT + col] = c[r] + bias;
  }
}

// ---------------------------------------------------------------------------
extern "C" void kernel_launch(void* const* d_in, const int* in_sizes, int n_in,
                              void* d_out, int out_size, void* d_ws, size_t ws_size,
                              hipStream_t stream) {
  (void)in_sizes; (void)n_in; (void)out_size; (void)ws_size;
  const float* x    = (const float*)d_in[0];  // [64, 512, 512]
  const float* Wih  = (const float*)d_in[1];  // [1024, 512]
  const float* b_ih = (const float*)d_in[2];  // [1024]
  const float* Whh  = (const float*)d_in[3];  // [1024, 1024]
  const float* b_hh = (const float*)d_in[4];  // [1024]
  const float* Who  = (const float*)d_in[5];  // [512, 1024]
  const float* b_ho = (const float*)d_in[6];  // [512]
  float* out = (float*)d_out;                 // [64, 512]

  // Workspace layout
  char* p = (char*)d_ws;
  float*  xproj  = (float*)p;   p += (size_t)NSEQ * NBATCH * DIM_HID * sizeof(float);
  __bf16* x_bf   = (__bf16*)p;  p += (size_t)NBATCH * NSEQ * DIM_IN * sizeof(__bf16);
  __bf16* wih_bf = (__bf16*)p;  p += (size_t)DIM_HID * DIM_IN * sizeof(__bf16);
  __bf16* whh_bf = (__bf16*)p;  p += (size_t)DIM_HID * DIM_HID * sizeof(__bf16);
  __bf16* who_bf = (__bf16*)p;  p += (size_t)DIM_OUT * DIM_HID * sizeof(__bf16);
  __bf16* h0     = (__bf16*)p;  p += (size_t)NBATCH * DIM_HID * sizeof(__bf16);
  __bf16* h1     = (__bf16*)p;  p += (size_t)NBATCH * DIM_HID * sizeof(__bf16);
  unsigned* bar  = (unsigned*)p;

  // fp32 -> bf16 staging passes (HBM-roofline streaming)
  auto conv = [&](const float* s, __bf16* d, size_t n) {
    int n8 = (int)(n / 8);
    k_f32_to_bf16v<<<(n8 + 255) / 256, 256, 0, stream>>>(s, d, n8);
  };
  conv(x,   x_bf,   (size_t)NBATCH * NSEQ * DIM_IN);
  conv(Wih, wih_bf, (size_t)DIM_HID * DIM_IN);
  conv(Whh, whh_bf, (size_t)DIM_HID * DIM_HID);
  conv(Who, who_bf, (size_t)DIM_OUT * DIM_HID);

  // h_{-1} = 0, barrier counter = 0 (graph-capture legal)
  (void)hipMemsetAsync(h0, 0, (size_t)NBATCH * DIM_HID * sizeof(__bf16), stream);
  (void)hipMemsetAsync(bar, 0, sizeof(unsigned), stream);

  // 1) x_proj = x @ W_ih^T + b_ih + b_hh   (stored [S, B, H])
  dim3 g1((NBATCH * NSEQ) / 128, DIM_HID / 64);
  k_xproj_gemm<<<g1, 256, 0, stream>>>(x_bf, wih_bf, b_ih, b_hh, xproj);

  // 2) persistent scan over 512 timesteps
  k_rnn_scan<<<64, 128, 0, stream>>>(xproj, whh_bf, h0, h1, bar);

  // 3) output projection of final hidden state (lands in h0 after 512 steps)
  k_out_proj<<<32, 128, 0, stream>>>(h0, who_bf, b_ho, out);
}
